// Decoder_68169720922393
// MI455X (gfx1250) — compile-verified, matrix operand order checked by reading, not run on hardware
//
#include <hip/hip_runtime.h>
#include <hip/hip_bf16.h>

typedef __attribute__((ext_vector_type(16))) __bf16 v16bf;
typedef __attribute__((ext_vector_type(8)))  __bf16 v8bf;
typedef __attribute__((ext_vector_type(8)))  float  v8f;
typedef __attribute__((ext_vector_type(4)))  int    v4i;

#define V_  32000
#define E_  512
#define H_  512
#define L_  4
#define B_  32
#define T_  51
#define HP_ (H_ + 8)   // padded LDS row: 520 elem = 1040 B = 260 dwords -> conflict-free

#define AS1 __attribute__((address_space(1)))
#define AS3 __attribute__((address_space(3)))

#if defined(__HIP_DEVICE_COMPILE__) && defined(__has_builtin)
#  if __has_builtin(__builtin_amdgcn_global_load_async_to_lds_b128)
#    define HAVE_ASYNC_LDS 1
#  else
#    define HAVE_ASYNC_LDS 0
#  endif
#  if __has_builtin(__builtin_amdgcn_s_wait_asynccnt)
#    define ASYNC_WAIT0() __builtin_amdgcn_s_wait_asynccnt(0)
#  else
#    define ASYNC_WAIT0() asm volatile("s_wait_asynccnt 0" ::: "memory")
#  endif
#else
#  define HAVE_ASYNC_LDS 0
#  define ASYNC_WAIT0()
#endif

// branchless activations: keep EXEC uniform between WMMA phases
__device__ __forceinline__ float fast_sigmoid(float x) {
    return __builtin_amdgcn_rcpf(1.0f + __expf(-x));
}
__device__ __forceinline__ float fast_tanh(float x) {
    return 1.0f - 2.0f * __builtin_amdgcn_rcpf(__expf(2.0f * x) + 1.0f);
}
__device__ __forceinline__ float clamp50_(float x) { return fminf(fmaxf(x, -50.0f), 50.0f); }

__device__ __forceinline__ v8f wmma_bf16(v8bf alo, v8bf ahi, v8bf blo, v8bf bhi, v8f c) {
    v16bf a = __builtin_shufflevector(alo, ahi, 0,1,2,3,4,5,6,7,8,9,10,11,12,13,14,15);
    v16bf b = __builtin_shufflevector(blo, bhi, 0,1,2,3,4,5,6,7,8,9,10,11,12,13,14,15);
    return __builtin_amdgcn_wmma_f32_16x16x32_bf16(false, a, false, b, (short)0, c, false, false);
}

// ---------------- prep: fp32 -> bf16 weight conversion + state copy ----------------
__global__ void prep_kernel(const float* __restrict__ wih, const float* __restrict__ whh,
                            const float* __restrict__ wout,
                            const float* __restrict__ h0, const float* __restrict__ c0,
                            __bf16* __restrict__ wih_b, __bf16* __restrict__ whh_b,
                            __bf16* __restrict__ wout_b,
                            float* __restrict__ h_st, float* __restrict__ c_st) {
    const size_t NW = (size_t)L_ * 4 * H_ * H_;   // 4,194,304
    const size_t NO = (size_t)V_ * H_;            // 16,384,000
    const size_t NS = (size_t)L_ * B_ * H_;       // 65,536
    const size_t total = 2 * NW + NO + 2 * NS;
    for (size_t i = (size_t)blockIdx.x * blockDim.x + threadIdx.x; i < total;
         i += (size_t)gridDim.x * blockDim.x) {
        if (i < NW)                    wih_b[i]               = (__bf16)wih[i];
        else if (i < 2 * NW)           whh_b[i - NW]          = (__bf16)whh[i - NW];
        else if (i < 2 * NW + NO)      wout_b[i - 2 * NW]     = (__bf16)wout[i - 2 * NW];
        else if (i < 2 * NW + NO + NS) h_st[i - 2 * NW - NO]  = h0[i - 2 * NW - NO];
        else                           c_st[i - 2 * NW - NO - NS] = c0[i - 2 * NW - NO - NS];
    }
}

// ---------------- one time step: embedding gather + 4 LSTM layers ----------------
// 1 workgroup, 512 threads = 16 waves; 64 (m,j) tile units, 4 per wave.
__global__ __launch_bounds__(512) void lstm_step_kernel(
    const float* __restrict__ emb, const int* __restrict__ target,
    const __bf16* __restrict__ wih, const __bf16* __restrict__ whh,
    const float* __restrict__ b_ih, const float* __restrict__ b_hh,
    float* __restrict__ h_st, float* __restrict__ c_st,
    __bf16* __restrict__ htop, int t)
{
    __shared__ __align__(16) __bf16 lds_x[B_][HP_];   // layer input (bf16)
    __shared__ __align__(16) __bf16 lds_h[B_][HP_];   // recurrent h (bf16)

    const int tid  = threadIdx.x;
    const int lane = tid & 31;
    const int wave = tid >> 5;
    const int l16  = (lane >> 4) & 1;   // lane >= 16
    const int lm   = lane & 15;
    const int kb   = l16 * 8;           // per-lane K base within a 32-K block

    // layer-0 input: embedding gather (teacher forcing: SOS at t==0 else target[:,t-1])
    for (int i = tid; i < B_ * H_; i += 512) {
        int b = i >> 9, jj = i & 511;
        int tok = (t == 0) ? 0 : target[b * T_ + (t - 1)];
        lds_x[b][jj] = (__bf16)emb[(size_t)tok * E_ + jj];
    }
    __syncthreads();

    for (int l = 0; l < L_; ++l) {
        const float* hsl = h_st + (size_t)l * B_ * H_;
        for (int i = tid; i < B_ * H_; i += 512) {        // clamped h from prev step
            int b = i >> 9, jj = i & 511;
            lds_h[b][jj] = (__bf16)hsl[(size_t)b * H_ + jj];
        }
        __syncthreads();

        const __bf16* Wih = wih + (size_t)l * 4 * H_ * H_;
        const __bf16* Whh = whh + (size_t)l * 4 * H_ * H_;
        const float*  bi  = b_ih + l * 4 * H_;
        const float*  bh  = b_hh + l * 4 * H_;

        float hpre[4][8];   // pre-clamp h, becomes next layer's x

        for (int u = 0; u < 4; ++u) {
            const int gu   = wave * 4 + u;
            const int m    = gu & 1;          // M tile (batch rows 0-15 / 16-31)
            const int j    = gu >> 1;         // N tile within H (0..31)
            const int jj   = j * 16 + lm;     // output column within H
            const int arow = m * 16 + lm;     // A-matrix row for this lane

            v8f acc[4];                       // i, f, g, o accumulators
#pragma unroll
            for (int g = 0; g < 4; ++g) {
                float bias = bi[g * H_ + jj] + bh[g * H_ + jj];
                acc[g] = (v8f){bias, bias, bias, bias, bias, bias, bias, bias};
            }

            for (int kk = 0; kk < 32; ++kk) { // K = 512 (x) + 512 (h), 32 per WMMA
                const __bf16* asrc;
                const __bf16* W;
                int k0;
                if (kk < 16) { asrc = &lds_x[arow][0]; W = Wih; k0 = kk * 32; }
                else         { asrc = &lds_h[arow][0]; W = Whh; k0 = (kk - 16) * 32; }
                v8bf alo = *(const v8bf*)(asrc + k0 + kb);
                v8bf ahi = *(const v8bf*)(asrc + k0 + 16 + kb);
#pragma unroll
                for (int g = 0; g < 4; ++g) {
                    const __bf16* wr = W + (size_t)(g * H_ + jj) * H_ + k0 + kb;
                    v8bf blo = *(const v8bf*)(wr);
                    v8bf bhi = *(const v8bf*)(wr + 16);
                    acc[g] = wmma_bf16(alo, ahi, blo, bhi, acc[g]);
                }
            }

            // pointwise LSTM cell: each (b, jj) owned by exactly one lane/VGPR
#pragma unroll
            for (int g8 = 0; g8 < 8; ++g8) {
                int brow   = m * 16 + g8 + l16 * 8;    // C/D layout: M = g + 8*(lane>=16)
                size_t idx = (size_t)l * B_ * H_ + (size_t)brow * H_ + jj;
                float iv = fast_sigmoid(acc[0][g8]);
                float fv = fast_sigmoid(acc[1][g8]);
                float gv = fast_tanh(acc[2][g8]);
                float ov = fast_sigmoid(acc[3][g8]);
                float cn = fv * c_st[idx] + iv * gv;
                float hn = ov * fast_tanh(cn);
                c_st[idx]   = clamp50_(cn);            // clamped carry
                h_st[idx]   = clamp50_(hn);            // clamped carry
                hpre[u][g8] = hn;                      // pre-clamp -> next layer input
            }
        }
        __syncthreads();   // all reads of lds_x done before overwrite
        for (int u = 0; u < 4; ++u) {
            const int gu = wave * 4 + u;
            const int m  = gu & 1;
            const int jj = (gu >> 1) * 16 + lm;
#pragma unroll
            for (int g8 = 0; g8 < 8; ++g8) {
                int brow = m * 16 + g8 + l16 * 8;
                __bf16 hb = (__bf16)hpre[u][g8];
                lds_x[brow][jj] = hb;
                if (l == L_ - 1) htop[(size_t)brow * H_ + jj] = hb;  // for projection
            }
        }
        __syncthreads();
    }
}

// ---------------- output projection: [32,512] x [512,32000] + b_out ----------------
// 250 blocks x 8 waves; A staged once per block in LDS (async), shared by 8 waves.
__global__ __launch_bounds__(256) void proj_kernel(
    const __bf16* __restrict__ htop, const __bf16* __restrict__ wout,
    const float* __restrict__ b_out, float* __restrict__ out, int t)
{
    __shared__ __align__(16) __bf16 lds_a[B_][HP_];

    const int tid  = threadIdx.x;
    const int lane = tid & 31;
    const int wave = tid >> 5;
    const int l16  = (lane >> 4) & 1;
    const int lm   = lane & 15;
    const int kb   = l16 * 8;
    const int v    = (blockIdx.x * 8 + wave) * 16 + lm;   // vocab column (< 32000)

    // stage htop [32,512] bf16 -> LDS: 2048 x 16B chunks, 8 per thread (no divergence)
#if HAVE_ASYNC_LDS
    for (int i = tid; i < (B_ * H_) / 8; i += 256) {
        int r = i >> 6, c = (i & 63) * 8;
        __builtin_amdgcn_global_load_async_to_lds_b128(
            (AS1 v4i*)(htop + (size_t)r * H_ + c),
            (AS3 v4i*)(&lds_a[r][c]), 0, 0);
    }
    ASYNC_WAIT0();
    __syncthreads();
#else
    for (int i = tid; i < (B_ * H_) / 8; i += 256) {
        int r = i >> 6, c = (i & 63) * 8;
        *(v8bf*)(&lds_a[r][c]) = *(const v8bf*)(htop + (size_t)r * H_ + c);
    }
    __syncthreads();
#endif

    float bias = b_out[v];
    v8f acc0 = (v8f){bias, bias, bias, bias, bias, bias, bias, bias};
    v8f acc1 = acc0;

    const __bf16* a0p = &lds_a[lm][0];        // batch rows 0..15
    const __bf16* a1p = &lds_a[16 + lm][0];   // batch rows 16..31
    const __bf16* bp  = wout + (size_t)v * H_;  // W_out row == B column (L2-resident)

#pragma unroll 4
    for (int k = 0; k < H_; k += 32) {
        v8bf blo  = *(const v8bf*)(bp  + k + kb);
        v8bf bhi  = *(const v8bf*)(bp  + k + 16 + kb);
        v8bf a0lo = *(const v8bf*)(a0p + k + kb);
        v8bf a0hi = *(const v8bf*)(a0p + k + 16 + kb);
        v8bf a1lo = *(const v8bf*)(a1p + k + kb);
        v8bf a1hi = *(const v8bf*)(a1p + k + 16 + kb);
        acc0 = wmma_bf16(a0lo, a0hi, blo, bhi, acc0);
        acc1 = wmma_bf16(a1lo, a1hi, blo, bhi, acc1);
    }

    const size_t tv = (size_t)t * V_ + v;
#pragma unroll
    for (int g8 = 0; g8 < 8; ++g8) {
        int b0 = g8 + l16 * 8;
        int b1 = 16 + g8 + l16 * 8;
        out[(size_t)b0 * T_ * V_ + tv] = acc0[g8];
        out[(size_t)b1 * T_ * V_ + tv] = acc1[g8];
    }
}

// ---------------- final hidden state -> tail of d_out ----------------
__global__ void hfin_kernel(const float* __restrict__ h_st, float* __restrict__ out) {
    int i = blockIdx.x * blockDim.x + threadIdx.x;
    if (i < L_ * B_ * H_) out[(size_t)B_ * T_ * V_ + i] = h_st[i];
}

extern "C" void kernel_launch(void* const* d_in, const int* in_sizes, int n_in,
                              void* d_out, int out_size, void* d_ws, size_t ws_size,
                              hipStream_t stream) {
    (void)in_sizes; (void)n_in; (void)out_size; (void)ws_size;
    // setup_inputs order: 0 encoder_output (unused), 1 decoder_hidden, 2 decoder_cell,
    // 3 target_tensor, 4 embedding, 5 W_ih, 6 W_hh, 7 b_ih, 8 b_hh, 9 W_out, 10 b_out
    const float* dec_h  = (const float*)d_in[1];
    const float* dec_c  = (const float*)d_in[2];
    const int*   target = (const int*)d_in[3];
    const float* emb    = (const float*)d_in[4];
    const float* W_ih   = (const float*)d_in[5];
    const float* W_hh   = (const float*)d_in[6];
    const float* b_ih   = (const float*)d_in[7];
    const float* b_hh   = (const float*)d_in[8];
    const float* W_out  = (const float*)d_in[9];
    const float* b_out  = (const float*)d_in[10];
    float* out = (float*)d_out;

    const size_t NW = (size_t)L_ * 4 * H_ * H_;   // weight elems per W_ih/W_hh
    const size_t NO = (size_t)V_ * H_;            // W_out elems
    const size_t NS = (size_t)L_ * B_ * H_;       // state elems

    char* ws = (char*)d_ws;
    size_t off = 0;
    auto take = [&](size_t bytes) -> char* {
        char* p = ws + off;
        off += (bytes + 255) & ~(size_t)255;
        return p;
    };
    __bf16* wih_b  = (__bf16*)take(NW * sizeof(__bf16));   //  8.4 MB
    __bf16* whh_b  = (__bf16*)take(NW * sizeof(__bf16));   //  8.4 MB
    __bf16* wout_b = (__bf16*)take(NO * sizeof(__bf16));   // 32.8 MB
    float*  h_st   = (float*) take(NS * sizeof(float));    // 256 KB
    float*  c_st   = (float*) take(NS * sizeof(float));    // 256 KB
    __bf16* htop   = (__bf16*)take((size_t)B_ * H_ * sizeof(__bf16)); // 32 KB

    prep_kernel<<<4096, 256, 0, stream>>>(W_ih, W_hh, W_out, dec_h, dec_c,
                                          wih_b, whh_b, wout_b, h_st, c_st);

    for (int t = 0; t < T_; ++t) {
        lstm_step_kernel<<<1, 512, 0, stream>>>(emb, target, wih_b, whh_b,
                                                b_ih, b_hh, h_st, c_st, htop, t);
        proj_kernel<<<V_ / 128, 256, 0, stream>>>(htop, wout_b, b_out, out, t);
    }

    hfin_kernel<<<(L_ * B_ * H_ + 255) / 256, 256, 0, stream>>>(h_st, out);
}
